// GATNRMS_49692771615099
// MI455X (gfx1250) — matrix operand order repaired and checked by reading, not compiled, not run on hardware
//
#include <hip/hip_runtime.h>

typedef __attribute__((ext_vector_type(16))) _Float16 v16h;
typedef __attribute__((ext_vector_type(8)))  float    v8f;
typedef __attribute__((ext_vector_type(4)))  float    f32x4;

#define BB 64
#define CC 5
#define HH 50
#define LL 32
#define DD 300
#define QQ 200
#define DP 320      // padded feature dim (5 strips of 64; K padded to mult of 32)
#define NSLOT 55    // C + H news slots
#define NN 2048     // rows per news GAT (B*L)
#define NU 3200     // rows in user GAT (B*H)

#define WMMA(A,B,C) __builtin_amdgcn_wmma_f32_16x16x32_f16(false,(A),false,(B),(short)0,(C),false,false)

#ifndef __has_builtin
#define __has_builtin(x) 0
#endif
#if __has_builtin(__builtin_amdgcn_tensor_load_to_lds) && __has_builtin(__builtin_amdgcn_s_wait_tensorcnt)
#define USE_TDM 1
#else
#define USE_TDM 0
#endif

#if USE_TDM
typedef unsigned int u32x4 __attribute__((ext_vector_type(4)));
typedef int i32x4 __attribute__((ext_vector_type(4)));
typedef int i32x8 __attribute__((ext_vector_type(8)));

// 1-D global->LDS DMA of n fp32 elements via the Tensor Data Mover.
__device__ __forceinline__ void tdm_load_1d_f32(unsigned ldsOff, const float* gsrc, int n) {
  unsigned long long ga = (unsigned long long)(size_t)gsrc;
  u32x4 g0;
  g0.x = 1u;                                                  // count=1, user mode
  g0.y = ldsOff;                                              // LDS byte address
  g0.z = (unsigned)(ga & 0xFFFFFFFFull);                      // global_addr[31:0]
  g0.w = (unsigned)((ga >> 32) & 0x01FFFFFFull) | (2u << 30); // addr[56:32] | type=2
  i32x8 g1;
  g1[0] = (int)(2u << 16);                                    // data_size = 4B
  g1[1] = (int)(((unsigned)n & 0xFFFFu) << 16);               // tensor_dim0[15:0]
  g1[2] = (int)((((unsigned)n >> 16) & 0xFFFFu) | (1u << 16));// tensor_dim0 hi | tensor_dim1=1
  g1[3] = (int)(((unsigned)n & 0xFFFFu) << 16);               // tile_dim0 = n
  g1[4] = 0;                                                  // tile_dim1/2 unused
  g1[5] = n;                                                  // tensor_dim0_stride
  g1[6] = 0;
  g1[7] = 0;
  i32x4 z4 = {0, 0, 0, 0};
#if __clang_major__ >= 23
  i32x8 z8 = {0, 0, 0, 0, 0, 0, 0, 0};
  __builtin_amdgcn_tensor_load_to_lds(g0, g1, z4, z4, z8, 0);
#else
  __builtin_amdgcn_tensor_load_to_lds(g0, g1, z4, z4, 0);
#endif
  __builtin_amdgcn_s_wait_tensorcnt(0);
}
#endif

// ---------------- K0: fp32 W -> f16, transposed + zero-padded: WT[n*DP+k] = W[k][n]
__global__ void k_convert_w(const float* __restrict__ W, _Float16* __restrict__ WT) {
  int idx = blockIdx.x * blockDim.x + threadIdx.x;
  if (idx >= DP * DP) return;
  int n = idx / DP, k = idx - n * DP;
  float v = (n < DD && k < DD) ? W[k * DD + n] : 0.0f;
  WT[idx] = (_Float16)v;
}

// ---------------- K1: Wh = X @ W via WMMA. 5 waves/block share one LDS A tile;
// wave w owns column strip w. Threads 0-127 each stage one aligned float4 piece
// of the tile from a register-resident gathered row pointer (300 % 4 == 0 so
// every piece is fully valid or fully zero padding).
__global__ void __launch_bounds__(160) k_wh_gemm(
    const float* __restrict__ src, const int* __restrict__ cand_t,
    const int* __restrict__ click_t, const _Float16* __restrict__ WT,
    _Float16* __restrict__ whc, int nrows, int mode) {
  __shared__ __align__(16) _Float16 As[16 * 32];
  const int tid  = threadIdx.x;
  const int lane = tid & 31, wave = tid >> 5;
  const int row0 = blockIdx.x * 16;
  const int slot = blockIdx.z;
  const int col0 = wave * 64;
  _Float16* __restrict__ wout = whc + (size_t)slot * DP * nrows;

  // staging assignment: thread tid<128 owns row r = tid>>3, k-piece kk4 = (tid&7)*4
  const int kk4 = (tid & 7) * 4;
  const float* rp = src;
  if (tid < 128) {
    int arow = row0 + (tid >> 3);
    if (mode == 0) {
      int b = arow >> 5, l = arow & 31;
      int title = (slot < CC) ? cand_t[(b * CC + slot) * LL + l]
                              : click_t[(b * HH + (slot - CC)) * LL + l];
      rp = src + (size_t)title * DD;
    } else {
      rp = src + (size_t)arow * DD;
    }
  }

  const int half = (lane < 16) ? 0 : 16;
  const int ko   = (lane < 16) ? 0 : 8;
  v8f acc0 = {}, acc1 = {}, acc2 = {}, acc3 = {};

  for (int kc = 0; kc < DP; kc += 32) {
    if (tid < 128) {
      const int k = kc + kk4;
      f32x4 x = {0.0f, 0.0f, 0.0f, 0.0f};
      if (k + 4 <= DD) x = *(const f32x4*)(rp + k);
      if (((tid & 7) == 0) && kc + 32 < DD) __builtin_prefetch(rp + kc + 32, 0, 1);
      _Float16* d = &As[(tid >> 3) * 32 + kk4];
      d[0] = (_Float16)x[0];
      d[1] = (_Float16)x[1];
      d[2] = (_Float16)x[2];
      d[3] = (_Float16)x[3];
    }
    __syncthreads();
    // A fragment: lanes 0-15 K{0..15}, lanes 16-31 K{16..31} (matches B)
    v16h a;
    {
      const _Float16* ar = &As[(lane & 15) * 32];
      #pragma unroll
      for (int t = 0; t < 8; ++t) { a[t] = ar[half + t]; a[8 + t] = ar[half + 8 + t]; }
    }
    // B fragments: WT column-major -> 16 consecutive K halves are contiguous 32B
    const _Float16* bbase = WT + (size_t)(col0 + (lane & 15)) * DP + kc + half;
    v16h b0 = *(const v16h*)(bbase);
    v16h b1 = *(const v16h*)(bbase + 16 * DP);
    v16h b2 = *(const v16h*)(bbase + 32 * DP);
    v16h b3 = *(const v16h*)(bbase + 48 * DP);
    acc0 = WMMA(a, b0, acc0);
    acc1 = WMMA(a, b1, acc1);
    acc2 = WMMA(a, b2, acc2);
    acc3 = WMMA(a, b3, acc3);
    __syncthreads();
  }
  // C layout: VGPR v -> M = v (+8 for lanes>=16), N = lane&15; store transposed f16
  const int rbase = row0 + ((lane < 16) ? 0 : 8);
  const int cb = col0 + (lane & 15);
  #pragma unroll
  for (int v = 0; v < 8; ++v) {
    wout[(size_t)(cb)      * nrows + rbase + v] = (_Float16)acc0[v];
    wout[(size_t)(cb + 16) * nrows + rbase + v] = (_Float16)acc1[v];
    wout[(size_t)(cb + 32) * nrows + rbase + v] = (_Float16)acc2[v];
    wout[(size_t)(cb + 48) * nrows + rbase + v] = (_Float16)acc3[v];
  }
}

// ---------------- K2: f_i = Wh_i . a[:D], g_i = Wh_i . a[D:]
__global__ void k_fg(const _Float16* __restrict__ whc, const float* __restrict__ avec,
                     float* __restrict__ f, float* __restrict__ g, int nrows, int nslots) {
  int idx = blockIdx.x * blockDim.x + threadIdx.x;
  if (idx >= nslots * nrows) return;
  int slot = idx / nrows, r = idx - slot * nrows;
  const _Float16* w = whc + (size_t)slot * DP * nrows + r;
  float sf = 0.0f, sg = 0.0f;
  for (int c = 0; c < DD; ++c) {
    float wv = (float)w[(size_t)c * nrows];
    sf = fmaf(wv, avec[c], sf);
    sg = fmaf(wv, avec[DD + c], sg);
  }
  f[idx] = sf;
  g[idx] = sg;
}

// ---------------- K3: per-slot softmax stats. lrelu monotone -> rowmax m_i = lrelu(f_i+gmax)
__global__ void __launch_bounds__(256) k_stats(
    const float* __restrict__ f, const float* __restrict__ g,
    float* __restrict__ m, float* __restrict__ rd, int nrows) {
  __shared__ __align__(16) float gs[NU];
  __shared__ float red[256];
  const int slot = blockIdx.x, tid = threadIdx.x;
  const float* fS = f + (size_t)slot * nrows;
  const float* gS = g + (size_t)slot * nrows;
  float lmax = -3.0e38f;
  for (int j = tid; j < nrows; j += 256) { float v = gS[j]; gs[j] = v; lmax = fmaxf(lmax, v); }
  red[tid] = lmax;
  __syncthreads();
  for (int s = 128; s > 0; s >>= 1) { if (tid < s) red[tid] = fmaxf(red[tid], red[tid + s]); __syncthreads(); }
  const float gmax = red[0];
  for (int i = tid; i < nrows; i += 256) {
    float fi = fS[i];
    float mi = fi + gmax; mi = (mi > 0.0f) ? mi : 0.2f * mi;
    float s = 0.0f;
    for (int j = 0; j < nrows; ++j) {
      float h = fi + gs[j];
      float e = (h > 0.0f) ? h : 0.2f * h;
      s += __expf(e - mi);
    }
    m[(size_t)slot * nrows + i]  = mi;
    rd[(size_t)slot * nrows + i] = 1.0f / s;
  }
}

// ---------------- K4: O = elu(att @ Wh). g vector DMA'd to LDS (TDM when available);
// att tiles synthesized cooperatively (4 exp per thread, straight-line) into LDS,
// 5 waves/block consume via WMMA.
__global__ void __launch_bounds__(160) k_att_gemm(
    const float* __restrict__ f, const float* __restrict__ g,
    const float* __restrict__ m, const float* __restrict__ rd,
    const _Float16* __restrict__ whc, _Float16* __restrict__ gat, int nrows) {
  __shared__ __align__(16) float    gsh[NU];
  __shared__ __align__(16) _Float16 Ash[16 * 32];
  const int tid  = threadIdx.x;
  const int lane = tid & 31, wave = tid >> 5;
  const int row0 = blockIdx.x * 16, slot = blockIdx.z;
  const int col0 = wave * 64;

  // synthesis assignment: thread tid<128 owns row r = tid>>3, j-piece kk4 = (tid&7)*4
  const int kk4 = (tid & 7) * 4;
  float fr = 0.0f, mr = 0.0f, rr = 0.0f;
  if (tid < 128) {
    int r = row0 + (tid >> 3);
    fr = f [(size_t)slot * nrows + r];
    mr = m [(size_t)slot * nrows + r];
    rr = rd[(size_t)slot * nrows + r];
  }
  const float* gS = g + (size_t)slot * nrows;
#if USE_TDM
  if (wave == 0) tdm_load_1d_f32((unsigned)(size_t)(&gsh[0]), gS, nrows);
#else
  for (int j = tid; j < nrows; j += 160) gsh[j] = gS[j];
#endif
  __syncthreads();

  const _Float16* wS = whc + (size_t)slot * DP * nrows;
  _Float16* oS = gat + (size_t)slot * nrows * DP;
  const int half16 = (lane < 16) ? 0 : 16;
  const int ko     = (lane < 16) ? 0 : 8;
  v8f acc0 = {}, acc1 = {}, acc2 = {}, acc3 = {};

  for (int jc = 0; jc < nrows; jc += 32) {
    if (tid < 128) {
      f32x4 gv = *(const f32x4*)(&gsh[jc + kk4]);
      _Float16* d = &Ash[(tid >> 3) * 32 + kk4];
      #pragma unroll
      for (int t = 0; t < 4; ++t) {
        float h = fr + gv[t];
        h = (h > 0.0f) ? h : 0.2f * h;
        d[t] = (_Float16)(__expf(h - mr) * rr);
      }
    }
    __syncthreads();
    v16h a;
    {
      const _Float16* ar = &Ash[(lane & 15) * 32];
      #pragma unroll
      for (int t = 0; t < 8; ++t) { a[t] = ar[half16 + t]; a[8 + t] = ar[half16 + 8 + t]; }
    }
    const _Float16* bbase = wS + (size_t)(col0 + (lane & 15)) * nrows + jc + half16;
    if (jc + 32 < nrows) __builtin_prefetch(bbase + 32, 0, 1);
    v16h b0 = *(const v16h*)(bbase);
    v16h b1 = *(const v16h*)(bbase + (size_t)16 * nrows);
    v16h b2 = *(const v16h*)(bbase + (size_t)32 * nrows);
    v16h b3 = *(const v16h*)(bbase + (size_t)48 * nrows);
    acc0 = WMMA(a, b0, acc0);
    acc1 = WMMA(a, b1, acc1);
    acc2 = WMMA(a, b2, acc2);
    acc3 = WMMA(a, b3, acc3);
    __syncthreads();
  }
  // fused ELU epilogue; store row-major f16 [nrows][DP]
  const int rbase = row0 + ((lane < 16) ? 0 : 8);
  const int cb = col0 + (lane & 15);
  #pragma unroll
  for (int v = 0; v < 8; ++v) {
    float x0 = acc0[v]; x0 = (x0 > 0.f) ? x0 : (__expf(x0) - 1.f);
    oS[(size_t)(rbase + v) * DP + cb]      = (_Float16)x0;
    float x1 = acc1[v]; x1 = (x1 > 0.f) ? x1 : (__expf(x1) - 1.f);
    oS[(size_t)(rbase + v) * DP + cb + 16] = (_Float16)x1;
    float x2 = acc2[v]; x2 = (x2 > 0.f) ? x2 : (__expf(x2) - 1.f);
    oS[(size_t)(rbase + v) * DP + cb + 32] = (_Float16)x2;
    float x3 = acc3[v]; x3 = (x3 > 0.f) ? x3 : (__expf(x3) - 1.f);
    oS[(size_t)(rbase + v) * DP + cb + 48] = (_Float16)x3;
  }
}

// ---------------- K5: additive attention (deterministic, no atomics)
__global__ void __launch_bounds__(256) k_addattn(
    const _Float16* __restrict__ gat, const float* __restrict__ lw,
    const float* __restrict__ lb, const float* __restrict__ qv,
    float* __restrict__ cand, float* __restrict__ click, float* __restrict__ user,
    int S, int mode) {
  __shared__ float xs[HH * 304];
  __shared__ float part[HH * 8];
  __shared__ float sc[HH];
  const int tid = threadIdx.x;
  const _Float16* xg; float* out;
  if (mode == 0) {
    int slot = blockIdx.x / BB, b = blockIdx.x - slot * BB;
    xg = gat + ((size_t)slot * NN + (size_t)b * LL) * DP;
    out = (slot < CC) ? cand + ((size_t)b * CC + slot) * DD
                      : click + ((size_t)b * HH + (slot - CC)) * DD;
  } else {
    int b = blockIdx.x;
    xg = gat + (size_t)b * HH * DP;
    out = user + (size_t)b * DD;
  }
  for (int i = tid; i < S * DD; i += 256) {
    int s = i / DD, c = i - s * DD;
    xs[s * 304 + c] = (float)xg[(size_t)s * DP + c];
  }
  __syncthreads();
  for (int p = tid; p < S * 8; p += 256) {
    int s = p >> 3, seg = p & 7;
    const float* xr = &xs[s * 304];
    float accv = 0.0f;
    for (int q = seg * 25; q < seg * 25 + 25; ++q) {
      const float* wr = lw + q * DD;
      float dot = lb[q];
      for (int c = 0; c < DD; ++c) dot = fmaf(xr[c], wr[c], dot);
      accv = fmaf(tanhf(dot), qv[q], accv);
    }
    part[p] = accv;
  }
  __syncthreads();
  if (tid < S) {
    float s0 = 0.0f;
    #pragma unroll
    for (int k = 0; k < 8; ++k) s0 += part[tid * 8 + k];
    sc[tid] = s0;
  }
  __syncthreads();
  if (tid == 0) {
    float mx = -3.0e38f;
    for (int s = 0; s < S; ++s) mx = fmaxf(mx, sc[s]);
    float sum = 0.0f;
    for (int s = 0; s < S; ++s) { float e = __expf(sc[s] - mx); sc[s] = e; sum += e; }
    float r = 1.0f / sum;
    for (int s = 0; s < S; ++s) sc[s] *= r;
  }
  __syncthreads();
  for (int c = tid; c < DD; c += 256) {
    float o = 0.0f;
    for (int s = 0; s < S; ++s) o = fmaf(sc[s], xs[s * 304 + c], o);
    out[c] = o;
  }
}

// ---------------- K6: out[b][c] = cand[b][c] . user[b]
__global__ void k_scores(const float* __restrict__ cand, const float* __restrict__ user,
                         float* __restrict__ out) {
  int idx = blockIdx.x * blockDim.x + threadIdx.x;
  if (idx >= BB * CC) return;
  int b = idx / CC;
  const float* cv = cand + (size_t)idx * DD;
  const float* uv = user + (size_t)b * DD;
  float s = 0.0f;
  for (int d = 0; d < DD; ++d) s = fmaf(cv[d], uv[d], s);
  out[idx] = s;
}

extern "C" void kernel_launch(void* const* d_in, const int* in_sizes, int n_in,
                              void* d_out, int out_size, void* d_ws, size_t ws_size,
                              hipStream_t stream) {
  (void)in_sizes; (void)n_in; (void)out_size; (void)ws_size;
  const int*   cand_t  = (const int*)  d_in[0];
  const int*   click_t = (const int*)  d_in[1];
  const float* emb     = (const float*)d_in[2];
  const float* Wn      = (const float*)d_in[3];
  const float* an      = (const float*)d_in[4];
  const float* lwn     = (const float*)d_in[5];
  const float* lbn     = (const float*)d_in[6];
  const float* qn      = (const float*)d_in[7];
  const float* Wu      = (const float*)d_in[8];
  const float* au      = (const float*)d_in[9];
  const float* lwu     = (const float*)d_in[10];
  const float* lbu     = (const float*)d_in[11];
  const float* qu      = (const float*)d_in[12];
  float* out = (float*)d_out;

  char* ws = (char*)d_ws;
  size_t off = 0;
  auto alloc = [&](size_t bytes) -> void* {
    void* p = ws + off;
    off = (off + bytes + 255) & ~(size_t)255;
    return p;
  };
  _Float16* W16n  = (_Float16*)alloc((size_t)DP * DP * 2);
  _Float16* W16u  = (_Float16*)alloc((size_t)DP * DP * 2);
  _Float16* whcN  = (_Float16*)alloc((size_t)NSLOT * DP * NN * 2);   // ~72 MB
  _Float16* whcU  = (_Float16*)alloc((size_t)DP * NU * 2);
  float*    fN    = (float*)alloc((size_t)NSLOT * NN * 4);
  float*    gN    = (float*)alloc((size_t)NSLOT * NN * 4);
  float*    mN    = (float*)alloc((size_t)NSLOT * NN * 4);
  float*    rN    = (float*)alloc((size_t)NSLOT * NN * 4);
  float*    fU    = (float*)alloc((size_t)NU * 4);
  float*    gU    = (float*)alloc((size_t)NU * 4);
  float*    mU    = (float*)alloc((size_t)NU * 4);
  float*    rU    = (float*)alloc((size_t)NU * 4);
  _Float16* gatN  = (_Float16*)alloc((size_t)NSLOT * NN * DP * 2);   // ~72 MB
  _Float16* gatU  = (_Float16*)alloc((size_t)NU * DP * 2);
  float*    candE = (float*)alloc((size_t)BB * CC * DD * 4);
  float*    clickE= (float*)alloc((size_t)BB * HH * DD * 4);
  float*    userV = (float*)alloc((size_t)BB * DD * 4);

  // weights -> f16 transposed/padded
  k_convert_w<<<(DP * DP + 255) / 256, 256, 0, stream>>>(Wn, W16n);
  k_convert_w<<<(DP * DP + 255) / 256, 256, 0, stream>>>(Wu, W16u);

  // ---- news encoder (55 slots per launch) ----
  k_wh_gemm<<<dim3(NN / 16, 1, NSLOT), 160, 0, stream>>>(
      emb, cand_t, click_t, W16n, whcN, NN, 0);
  k_fg<<<(NSLOT * NN + 255) / 256, 256, 0, stream>>>(whcN, an, fN, gN, NN, NSLOT);
  k_stats<<<NSLOT, 256, 0, stream>>>(fN, gN, mN, rN, NN);
  k_att_gemm<<<dim3(NN / 16, 1, NSLOT), 160, 0, stream>>>(
      fN, gN, mN, rN, whcN, gatN, NN);
  k_addattn<<<NSLOT * BB, 256, 0, stream>>>(
      gatN, lwn, lbn, qn, candE, clickE, nullptr, LL, 0);

  // ---- user encoder ----
  k_wh_gemm<<<dim3(NU / 16, 1, 1), 160, 0, stream>>>(
      clickE, nullptr, nullptr, W16u, whcU, NU, 1);
  k_fg<<<(NU + 255) / 256, 256, 0, stream>>>(whcU, au, fU, gU, NU, 1);
  k_stats<<<1, 256, 0, stream>>>(fU, gU, mU, rU, NU);
  k_att_gemm<<<dim3(NU / 16, 1, 1), 160, 0, stream>>>(
      fU, gU, mU, rU, whcU, gatU, NU);
  k_addattn<<<BB, 256, 0, stream>>>(gatU, lwu, lbu, qu, nullptr, nullptr, userV, HH, 1);

  // ---- click predictor ----
  k_scores<<<(BB * CC + 255) / 256, 256, 0, stream>>>(candE, userV, out);
}
